// Conv2dQInt8ModuleBase_81552839016699
// MI455X (gfx1250) — 8-bit matrix operand order repaired and checked by reading, compiled, not run on hardware
//
#include <hip/hip_runtime.h>
#include <hip/hip_bf16.h>


typedef int v8i __attribute__((ext_vector_type(8)));

#define IN_ZP 7
#define W_ZP 3
#define ZPK (IN_ZP * W_ZP * 576)
#define SCALE 1.0e-4f

#define OW 254
#define OH 254
#define IW 256
#define IH 256
#define NC 64

#define TILE_PX 64
#define RAW_XS 68
#define RAW_RSTRIDE (RAW_XS * NC)
#define BCOL 592
#define B_BYTES (64 * BCOL)

__global__ void qconv_pack_w(const int* __restrict__ w,
                             signed char* __restrict__ wB,
                             int* __restrict__ colsum) {
  int o = threadIdx.x;
  if (o >= 64) return;
  int sum = 0;
  for (int c = 0; c < 64; ++c) {
    for (int r = 0; r < 3; ++r) {
      for (int s = 0; s < 3; ++s) {
        int v = w[((o * 64 + c) * 3 + r) * 3 + s];
        sum += v;
        int k = (r * 3 + s) * 64 + c;
        wB[o * BCOL + k] = (signed char)v;
      }
    }
  }
  colsum[o] = sum;
}

__global__ __launch_bounds__(256, 2)
void qconv_main(const int* __restrict__ in,
                const signed char* __restrict__ wB,
                const int* __restrict__ colsum,
                const int* __restrict__ bias,
                int* __restrict__ out) {
  __shared__ __align__(16) signed char rawA[3 * RAW_RSTRIDE];
  __shared__ __align__(16) signed char bT[B_BYTES];

  const int tid = threadIdx.x;
  const int tileId = blockIdx.x;
  const int tx = tileId & 3;
  const int rest = tileId >> 2;
  const int oy = rest % OH;
  const int nimg = rest / OH;
  const int ox0 = tx * TILE_PX;

  const int inNBase = nimg * NC * IH * IW;
  for (int idx = tid; idx < 64 * 3 * 17; idx += 256) {
    int c = idx / 51;
    int rem = idx - c * 51;
    int r = rem / 17;
    int xg = rem - r * 17;
    int x0 = xg * 4;
    const int* src = in + inNBase + (c * IH + (oy + r)) * IW;
    int gx0 = ox0 + x0;
    int4 v;
    if (gx0 + 3 <= IW - 1) {
      v = *(const int4*)(src + gx0);
    } else {
      v.x = src[min(gx0 + 0, IW - 1)];
      v.y = src[min(gx0 + 1, IW - 1)];
      v.z = src[min(gx0 + 2, IW - 1)];
      v.w = src[min(gx0 + 3, IW - 1)];
    }
    signed char* dst = &rawA[r * RAW_RSTRIDE + x0 * NC + c];
    dst[0 * NC] = (signed char)v.x;
    dst[1 * NC] = (signed char)v.y;
    dst[2 * NC] = (signed char)v.z;
    dst[3 * NC] = (signed char)v.w;
  }

  for (int idx = tid; idx < B_BYTES / 16; idx += 256) {
    ((int4*)bT)[idx] = ((const int4*)wB)[idx];
  }

  __syncthreads();

  const int wave = tid >> 5;
  const int lane = tid & 31;
  const int mt = wave & 3;
  const int nt0 = (wave >> 2) * 2;
  const int lcol = lane & 15;
  const int hi = (lane & 16) ? 1 : 0;
  const int aHalf = hi ? 8 : 0;
  const int bHalf = hi ? 16 : 0;

  const int o0 = nt0 * 16 + lcol;
  const int o1 = o0 + 16;
  const int m_lane = mt * 16 + lcol;

  v8i acc0 = {};
  v8i acc1 = {};
  v8i accS = {};
  v8i onesB;
#pragma unroll
  for (int i = 0; i < 8; ++i) onesB[i] = 0x01010101;

#pragma unroll
  for (int rs = 0; rs < 9; ++rs) {
    const int r = rs / 3;
    const int s = rs - r * 3;

    const signed char* ap = &rawA[r * RAW_RSTRIDE + (m_lane + s) * NC + aHalf];
    int4 t0 = *(const int4*)(ap - aHalf + bHalf + 0);
    int4 t1 = *(const int4*)(ap - aHalf + bHalf + 32);
    v8i a;
    a[0] = t0.x; a[1] = t0.y; a[2] = t0.z; a[3] = t0.w;
    a[4] = t1.x; a[5] = t1.y; a[6] = t1.z; a[7] = t1.w;

    const signed char* bp0 = &bT[o0 * BCOL + rs * 64 + bHalf];
    int4 u0 = *(const int4*)(bp0 + 0);
    int4 u1 = *(const int4*)(bp0 + 32);
    v8i b0;
    b0[0] = u0.x; b0[1] = u0.y; b0[2] = u0.z; b0[3] = u0.w;
    b0[4] = u1.x; b0[5] = u1.y; b0[6] = u1.z; b0[7] = u1.w;

    const signed char* bp1 = &bT[o1 * BCOL + rs * 64 + bHalf];
    int4 u2 = *(const int4*)(bp1 + 0);
    int4 u3 = *(const int4*)(bp1 + 32);
    v8i b1;
    b1[0] = u2.x; b1[1] = u2.y; b1[2] = u2.z; b1[3] = u2.w;
    b1[4] = u3.x; b1[5] = u3.y; b1[6] = u3.z; b1[7] = u3.w;

    acc0 = __builtin_amdgcn_wmma_i32_16x16x64_iu8(true, a, true, b0, acc0, false, false);
    acc1 = __builtin_amdgcn_wmma_i32_16x16x64_iu8(true, a, true, b1, acc1, false, false);
    accS = __builtin_amdgcn_wmma_i32_16x16x64_iu8(true, a, true, onesB, accS, false, false);
  }

  const int cs0 = ZPK - W_ZP   * 0 - IN_ZP * colsum[o0];
  const int cs1 = ZPK - IN_ZP * colsum[o1];
  const float bf0 = (float)bias[o0];
  const float bf1 = (float)bias[o1];
  const int oxBase = ox0 + mt * 16 + aHalf;

  auto emit = [&](const v8i& acc, int kfix, float bf, int o) {
    int q[8];
#pragma unroll
    for (int j = 0; j < 8; ++j) {
      int adj = acc[j] - W_ZP * accS[j] + kfix;
      float v = fmaf((float)adj, SCALE, bf);
      q[j] = __float2int_rn(v);
    }
    int base = ((nimg * 64 + o) * OH + oy) * OW;
    if (oxBase + 7 < OW) {
      int2* dst = (int2*)(out + base + oxBase);
      dst[0] = make_int2(q[0], q[1]);
      dst[1] = make_int2(q[2], q[3]);
      dst[2] = make_int2(q[4], q[5]);
      dst[3] = make_int2(q[6], q[7]);
    } else {
#pragma unroll
      for (int j = 0; j < 8; ++j) {
        if (oxBase + j < OW) out[base + oxBase + j] = q[j];
      }
    }
  };

  emit(acc0, cs0, bf0, o0);
  emit(acc1, cs1, bf1, o1);
}

extern "C" void kernel_launch(void* const* d_in, const int* in_sizes, int n_in,
                              void* d_out, int out_size, void* d_ws, size_t ws_size,
                              hipStream_t stream) {
  const int* input  = (const int*)d_in[0];
  const int* weight = (const int*)d_in[1];
  const int* bias   = (const int*)d_in[2];
  int* out = (int*)d_out;

  signed char* wsB = (signed char*)d_ws;
  int* wsColsum = (int*)((char*)d_ws + B_BYTES);

  qconv_pack_w<<<1, 64, 0, stream>>>(weight, wsB, wsColsum);

  const int numTiles = 16 * OH * 4;
  qconv_main<<<numTiles, 256, 0, stream>>>(input, wsB, wsColsum, bias, out);
}
